// MultiHeadCausalAttention_22557168238663
// MI455X (gfx1250) — matrix operand order repaired and checked by reading, not compiled, not run on hardware
//
#include <hip/hip_runtime.h>
#include <hip/hip_bf16.h>

#define S_LEN 2048
#define D_IN  1024
#define NH    16
#define EH    64

typedef __attribute__((ext_vector_type(16))) _Float16 v16h;
typedef __attribute__((ext_vector_type(8)))  _Float16 h8;
typedef __attribute__((ext_vector_type(8)))  float    v8f;

union Frag { v16h v; h8 h[2]; };

static __device__ __forceinline__ v8f wmma16(v16h a, v16h b, v8f c) {
    // D = A(16x32 f16) * B(32x16 f16) + C(16x16 f32)
    return __builtin_amdgcn_wmma_f32_16x16x32_f16(
        /*neg_a=*/false, a, /*neg_b=*/false, b,
        /*c_mod=*/(short)0, c, /*reuse_a=*/false, /*reuse_b=*/false);
}

// ---------------- prep: fp32 -> f16 conversions ----------------

__global__ void k_cvt_x(const float* __restrict__ x, _Float16* __restrict__ xh, int n) {
    int i = blockIdx.x * blockDim.x + threadIdx.x;
    if (i < n) xh[i] = (_Float16)x[i];
}

// Wt[mat][h][e][d] = W[mat][h][d][e]  (f16): B-fragments then read contiguously in d
__global__ void k_tr_w(const float* __restrict__ wq, const float* __restrict__ wk,
                       const float* __restrict__ wv, _Float16* __restrict__ wt) {
    const int per = NH * EH * D_IN;
    int idx = blockIdx.x * blockDim.x + threadIdx.x;
    if (idx >= 3 * per) return;
    int mat = idx / per;
    int r   = idx - mat * per;
    int h   = r / (EH * D_IN);
    int r2  = r - h * (EH * D_IN);
    int e   = r2 / D_IN;
    int d   = r2 - e * D_IN;
    const float* w = (mat == 0) ? wq : (mat == 1) ? wk : wv;
    wt[idx] = (_Float16)w[h * (D_IN * EH) + d * EH + e];
}

// ---------------- fused QKV projection ----------------
// gridDim = (S/64, H, 3); one wave computes a 64x64 tile of one head's q/k/v.
// __launch_bounds__(32,1): budget registers for 1 wave/EU so the 16 f32
// accumulators (128 VGPRs) + fragments stay register-resident (no scratch).

__global__ __launch_bounds__(32, 1)
void k_qkv(const _Float16* __restrict__ xh,
           const _Float16* __restrict__ wt_all,  // [3][H][E][D]
           const float* __restrict__ bq, const float* __restrict__ bk,
           const float* __restrict__ bv,
           _Float16* __restrict__ qo, _Float16* __restrict__ ko,
           _Float16* __restrict__ vo)
{
    const int lane = threadIdx.x & 31;
    const int sub  = lane >> 4;      // half-wave select
    const int l16  = lane & 15;
    const int m0   = blockIdx.x * 64;
    const int h    = blockIdx.y;
    const int mat  = blockIdx.z;

    const _Float16* wb   = wt_all + ((size_t)mat * NH + h) * (EH * D_IN);
    const float*    bias = (mat == 0) ? bq : (mat == 1) ? bk : bv;

    v8f zero = {};
    v8f acc[4][4];
#pragma unroll
    for (int mt = 0; mt < 4; ++mt)
#pragma unroll
        for (int nt = 0; nt < 4; ++nt) acc[mt][nt] = zero;

    for (int k0 = 0; k0 < D_IN; k0 += 32) {
        Frag a[4];   // A: rows of x, K-chunks per ISA 16-bit A layout
#pragma unroll
        for (int mt = 0; mt < 4; ++mt) {
            const _Float16* xr = xh + (m0 + mt * 16 + l16) * D_IN + k0;
            a[mt].h[0] = *(const h8*)(xr + sub * 8);
            a[mt].h[1] = *(const h8*)(xr + 16 + sub * 8);
        }
#pragma unroll
        for (int nt = 0; nt < 4; ++nt) {
            Frag b;  // B: column e = nt*16+l16, K-chunks per ISA 16-bit layout (sub*8, 16+sub*8)
            const _Float16* wr = wb + (nt * 16 + l16) * D_IN + k0 + sub * 8;
            b.h[0] = *(const h8*)(wr);
            b.h[1] = *(const h8*)(wr + 16);
#pragma unroll
            for (int mt = 0; mt < 4; ++mt)
                acc[mt][nt] = wmma16(a[mt].v, b.v, acc[mt][nt]);
        }
    }

#pragma unroll
    for (int nt = 0; nt < 4; ++nt) {
        const int e = nt * 16 + l16;
        const float bvv = bias[h * EH + e];
#pragma unroll
        for (int mt = 0; mt < 4; ++mt) {
            if (mat == 2) {
                // v^T: lane holds fixed e, 8 consecutive s -> one b128 store
                h8 pk;
#pragma unroll
                for (int i = 0; i < 8; ++i) pk[i] = (_Float16)(acc[mt][nt][i] + bvv);
                *(h8*)(vo + h * (EH * S_LEN) + e * S_LEN + m0 + mt * 16 + sub * 8) = pk;
            } else {
                _Float16* outp = (mat == 0) ? qo : ko;
#pragma unroll
                for (int i = 0; i < 8; ++i) {
                    const int srow = m0 + mt * 16 + sub * 8 + i;
                    outp[h * (S_LEN * EH) + srow * EH + e] = (_Float16)(acc[mt][nt][i] + bvv);
                }
            }
        }
    }
}

// ---------------- causal flash attention: one wave = 32 queries of one head ----------------
// S^T = K * Q^T tiles (K/V fragments shared across both query tiles);
// online softmax along keys; ctx^T = V^T * P accumulated in registers.
// __launch_bounds__(32,1): keep O/Q/K/V fragments (~170 VGPRs) register-resident.

__global__ __launch_bounds__(32, 1)
void k_attn(const _Float16* __restrict__ qh,   // [H][S][E]
            const _Float16* __restrict__ kh,   // [H][S][E]
            const _Float16* __restrict__ vt,   // [H][E][S]
            float* __restrict__ out)           // [S][H*E]
{
    const int lane = threadIdx.x & 31;
    const int sub  = lane >> 4;
    const int l16  = lane & 15;
    const int q0   = blockIdx.x * 32;
    const int h    = blockIdx.y;

    int qg[2];
    qg[0] = q0 + l16;
    qg[1] = q0 + 16 + l16;

    // Q^T B-fragments: [q-tile][e-half]; lane = column q, e = sub*8+{0..7}, 16+sub*8+{0..7}
    Frag qb[2][2];
#pragma unroll
    for (int qt = 0; qt < 2; ++qt) {
        const _Float16* qrow = qh + h * (S_LEN * EH) + (q0 + qt * 16 + l16) * EH;
        qb[qt][0].h[0] = *(const h8*)(qrow + sub * 8);
        qb[qt][0].h[1] = *(const h8*)(qrow + 16 + sub * 8);
        qb[qt][1].h[0] = *(const h8*)(qrow + 32 + sub * 8);
        qb[qt][1].h[1] = *(const h8*)(qrow + 48 + sub * 8);
    }

    v8f zero = {};
    v8f O[2][4];
#pragma unroll
    for (int qt = 0; qt < 2; ++qt)
#pragma unroll
        for (int eo = 0; eo < 4; ++eo) O[qt][eo] = zero;
    float mrun[2] = { -__builtin_inff(), -__builtin_inff() };
    float lrun[2] = { 0.f, 0.f };

    const _Float16* kbase = kh + h * (S_LEN * EH);
    const _Float16* vbase = vt + h * (EH * S_LEN);
    const float sc = 0.125f;   // 1/sqrt(64)

    const int nsteps = q0 / 32 + 1;   // 32-key steps covering keys 0..q0+31
    for (int st = 0; st < nsteps; ++st) {
        const int s0 = st * 32;

        // ---- K A-fragments: [key-tile][e-half], shared by both query tiles ----
        const _Float16* kr0 = kbase + (s0 + l16) * EH;
        const _Float16* kr1 = kbase + (s0 + 16 + l16) * EH;
        Frag ka[2][2];
        ka[0][0].h[0] = *(const h8*)(kr0 + sub * 8);
        ka[0][0].h[1] = *(const h8*)(kr0 + 16 + sub * 8);
        ka[0][1].h[0] = *(const h8*)(kr0 + 32 + sub * 8);
        ka[0][1].h[1] = *(const h8*)(kr0 + 48 + sub * 8);
        ka[1][0].h[0] = *(const h8*)(kr1 + sub * 8);
        ka[1][0].h[1] = *(const h8*)(kr1 + 16 + sub * 8);
        ka[1][1].h[0] = *(const h8*)(kr1 + 32 + sub * 8);
        ka[1][1].h[1] = *(const h8*)(kr1 + 48 + sub * 8);
        __builtin_prefetch(kr0 + 32 * EH, 0, 3);   // next key block
        __builtin_prefetch(kr1 + 32 * EH, 0, 3);

        // ---- V A-fragments: [e-tile], shared by both query tiles ----
        Frag va[4];
#pragma unroll
        for (int eo = 0; eo < 4; ++eo) {
            const _Float16* vr = vbase + (eo * 16 + l16) * S_LEN + s0;
            va[eo].h[0] = *(const h8*)(vr + sub * 16);
            va[eo].h[1] = *(const h8*)(vr + sub * 16 + 8);
        }

#pragma unroll
        for (int qt = 0; qt < 2; ++qt) {
            // ---- scores: two 16x16 tiles of S^T for this query tile ----
            v8f sf0 = zero, sf1 = zero;
            sf0 = wmma16(ka[0][0].v, qb[qt][0].v, sf0);
            sf0 = wmma16(ka[0][1].v, qb[qt][1].v, sf0);
            sf1 = wmma16(ka[1][0].v, qb[qt][0].v, sf1);
            sf1 = wmma16(ka[1][1].v, qb[qt][1].v, sf1);

            // ---- causal mask + online softmax (per query column) ----
            const int qc = qg[qt];
            float p0[8], p1[8];
            float cmax = -3.0e38f;
#pragma unroll
            for (int i = 0; i < 8; ++i) {
                const int sk0 = s0 + sub * 8 + i;        // C layout: row = (lane/16)*8+i
                const int sk1 = s0 + 16 + sub * 8 + i;
                float v0 = (sk0 <= qc) ? sf0[i] * sc : -__builtin_inff();
                float v1 = (sk1 <= qc) ? sf1[i] * sc : -__builtin_inff();
                p0[i] = v0; p1[i] = v1;
                cmax = fmaxf(cmax, fmaxf(v0, v1));
            }
            cmax = fmaxf(cmax, __shfl_xor(cmax, 16, 32));
            const float mnew  = fmaxf(mrun[qt], cmax);
            const float alpha = __expf(mrun[qt] - mnew);
            float rsum = 0.f;
#pragma unroll
            for (int i = 0; i < 8; ++i) {
                p0[i] = __expf(p0[i] - mnew);
                p1[i] = __expf(p1[i] - mnew);
                rsum += p0[i] + p1[i];
            }
            rsum += __shfl_xor(rsum, 16, 32);
            lrun[qt] = lrun[qt] * alpha + rsum;
            mrun[qt] = mnew;
#pragma unroll
            for (int eo = 0; eo < 4; ++eo) O[qt][eo] = O[qt][eo] * alpha;

            // ---- reshape P from C layout to f16 B-fragment (K = 32 keys) ----
            Frag pb;
#pragma unroll
            for (int i = 0; i < 8; ++i) {
                const float o0 = __shfl_xor(p0[i], 16, 32);
                const float o1 = __shfl_xor(p1[i], 16, 32);
                pb.v[i]     = (_Float16)(sub ? o1 : p0[i]);   // B rows sub*16 + {0..7}
                pb.v[i + 8] = (_Float16)(sub ? p1[i] : o0);   // B rows sub*16 + {8..15}
            }

            // ---- ctx^T += V^T tile * P ----
#pragma unroll
            for (int eo = 0; eo < 4; ++eo)
                O[qt][eo] = wmma16(va[eo].v, pb.v, O[qt][eo]);
        }
    }

#pragma unroll
    for (int qt = 0; qt < 2; ++qt) {
        const float linv = 1.0f / lrun[qt];
#pragma unroll
        for (int eo = 0; eo < 4; ++eo)
#pragma unroll
            for (int i = 0; i < 8; ++i) {
                const int e = eo * 16 + sub * 8 + i;
                out[qg[qt] * (NH * EH) + h * EH + e] = O[qt][eo][i] * linv;
            }
    }
}

// ---------------- host launch ----------------

extern "C" void kernel_launch(void* const* d_in, const int* in_sizes, int n_in,
                              void* d_out, int out_size, void* d_ws, size_t ws_size,
                              hipStream_t stream) {
    const float* x  = (const float*)d_in[0];
    const float* Wq = (const float*)d_in[1];
    const float* Wk = (const float*)d_in[2];
    const float* Wv = (const float*)d_in[3];
    const float* bq = (const float*)d_in[4];
    const float* bk = (const float*)d_in[5];
    const float* bv = (const float*)d_in[6];
    float* out = (float*)d_out;

    // workspace (f16): xh[S*D] | Wt[3][H][E][D] | q[H][S][E] | k[H][S][E] | vT[H][E][S]
    char* ws = (char*)d_ws;
    _Float16* xh  = (_Float16*)ws;
    _Float16* wt  = xh  + (size_t)S_LEN * D_IN;
    _Float16* qhb = wt  + (size_t)3 * NH * EH * D_IN;
    _Float16* khb = qhb + (size_t)NH * S_LEN * EH;
    _Float16* vtb = khb + (size_t)NH * S_LEN * EH;

    k_cvt_x<<<(S_LEN * D_IN + 255) / 256, 256, 0, stream>>>(x, xh, S_LEN * D_IN);
    k_tr_w<<<(3 * NH * EH * D_IN + 255) / 256, 256, 0, stream>>>(Wq, Wk, Wv, wt);

    k_qkv<<<dim3(S_LEN / 64, NH, 3), 32, 0, stream>>>(xh, wt, bq, bk, bv, qhb, khb, vtb);

    k_attn<<<dim3(S_LEN / 32, NH), 32, 0, stream>>>(qhb, khb, vtb, out);
}
